// AttNet_17686675324979
// MI455X (gfx1250) — matrix operand order repaired and checked by reading, not compiled, not run on hardware
//
#include <hip/hip_runtime.h>
#include <hip/hip_bf16.h>
#include <math.h>
#include <stdint.h>

typedef __attribute__((ext_vector_type(16))) _Float16 v16h;
typedef __attribute__((ext_vector_type(8)))  float    v8f;

#define BATCH 32
#define COUT 64

// ---------------------------------------------------------------------------
// Weight packing: f32 [64, CIN, 3, 3] -> f16 WMMA A-fragments.
// A-matrix 16x32 f16 layout (ISA 7.12.2): lane l: M=l%16, group=l/16,
// VGPR v holds halves (2v,2v+1) with K = (v<4 ? 2v : 16+2(v-4)) + 8*group + parity
// Packed index: ((mb*KST + ks)*32 + lane)*16 + h   (contiguous 32B per lane)
// ---------------------------------------------------------------------------
template <int CIN>
__global__ void pack_w_kernel(const float* __restrict__ w, _Float16* __restrict__ packA) {
    constexpr int K_TOT = CIN * 9;
    constexpr int KST = (K_TOT + 31) / 32;
    const int total = 4 * KST * 32 * 16;
    int idx = blockIdx.x * 256 + threadIdx.x;
    if (idx >= total) return;
    int h    = idx & 15;
    int lane = (idx >> 4) & 31;
    int ks   = (idx >> 9) % KST;
    int mb   = idx / (KST * 512);
    int m = lane & 15, grp = lane >> 4;
    int v = h >> 1, par = h & 1;
    int koff = ((v < 4) ? (v * 2) : (16 + (v - 4) * 2)) + grp * 8 + par;
    int k = ks * 32 + koff;
    _Float16 val = (_Float16)0.0f;
    if (k < K_TOT) {
        int ci = k / 9, r = k - ci * 9;
        val = (_Float16)w[((mb * 16 + m) * CIN + ci) * 9 + r];
    }
    packA[idx] = val;
}

// ---------------------------------------------------------------------------
// Fused conv3x3 + bias (+BN +ReLU) (+2x2 maxpool), implicit GEMM with WMMA.
// WG = 256 threads = 8 waves. GEMM tile: M=64 couts x N=64 conv pixels
// (8x8 conv patch -> pools to 4x4). Wave w: wm=w&3 selects 16 couts,
// wn=w>>2 selects 32 pixels; each wave runs 2 WMMA tiles (2 accumulators).
//
// Pipeline: global --async--> LDS raw tile (interior fast path) ->
//           LDS im2col B-pack (fragment order, f32->f16) ->
//           32B vector LDS reads per fragment -> v_wmma_f32_16x16x32_f16.
// lds_out aliases lds_in (dead after stage 2) to keep LDS ~99KB -> 3 WG/WGP.
// ---------------------------------------------------------------------------
template <int CIN, int PAD, bool POOL, bool BNRELU>
__global__ void conv_block_kernel(const float* __restrict__ x,
                                  const _Float16* __restrict__ packA,
                                  const float* __restrict__ cb,
                                  const float* __restrict__ g,
                                  const float* __restrict__ be,
                                  float* __restrict__ y,
                                  int H, int W) {
    constexpr int K_TOT = CIN * 9;
    constexpr int KST = (K_TOT + 31) / 32;
    constexpr int INOUT_F = (CIN * 100 > COUT * 64) ? (CIN * 100) : (COUT * 64);
    const int Hc = H + 2 * PAD - 2, Wc = W + 2 * PAD - 2;
    const int nbw = (Wc + 7) / 8, nbh = (Hc + 7) / 8;

    int bid = blockIdx.x;
    int b   = bid / (nbh * nbw);
    int rem = bid % (nbh * nbw);
    int oh0 = (rem / nbw) * 8;
    int ow0 = (rem % nbw) * 8;

    __shared__ __align__(32) char smem[INOUT_F * 4 + KST * 2048 * 2];
    float*    lds_in    = (float*)smem;                   // [CIN][10][10] (stage 1-2)
    float*    lds_out   = (float*)smem;                   // [cout][64] (aliases, stage 3-4)
    _Float16* lds_bpack = (_Float16*)(smem + INOUT_F * 4);// [ks][grp][64 pixel][16 halves]

    const int tid = threadIdx.x;

    // ---- Stage 1: raw input tile to LDS ----
    const bool interior = (oh0 - PAD >= 0) && (oh0 - PAD + 10 <= H) &&
                          (ow0 - PAD >= 0) && (ow0 - PAD + 10 <= W);
    if (interior) {
        // Fast path: CDNA5 async global->LDS copy, no VGPR round-trip.
        const float* src = x + ((size_t)(b * CIN) * H + (oh0 - PAD)) * W + (ow0 - PAD);
        for (int i = tid; i < CIN * 100; i += 256) {
            int ci = i / 100;
            int r2 = i - ci * 100;
            int rr = r2 / 10, cc = r2 - rr * 10;
            unsigned ldsa = (unsigned)(uintptr_t)(&lds_in[i]);  // addr[31:0] = LDS offset
            unsigned long long ga =
                (unsigned long long)(uintptr_t)(src + ((size_t)ci * H + rr) * W + cc);
            asm volatile("global_load_async_to_lds_b32 %0, %1, off"
                         :: "v"(ldsa), "v"(ga) : "memory");
        }
        asm volatile("s_wait_asynccnt 0x0" ::: "memory");
    } else {
        // Boundary path: zero-padded scalar staging.
        for (int i = tid; i < CIN * 100; i += 256) {
            int ci = i / 100;
            int r2 = i - ci * 100;
            int rr = r2 / 10, cc = r2 - rr * 10;
            int ih = oh0 - PAD + rr, iw = ow0 - PAD + cc;
            float v = 0.0f;
            if (ih >= 0 && ih < H && iw >= 0 && iw < W)
                v = x[((size_t)(b * CIN + ci) * H + ih) * W + iw];
            lds_in[i] = v;
        }
    }
    __syncthreads();

    // ---- Stage 2: im2col pack into WMMA-B fragment order (once per WG) ----
    // j = ((ks*2 + grp)*64 + pixel)*16 + h ; K index = ks*32 + grp*8 + (h<8 ? h : h+8)
    for (int j = tid; j < KST * 2048; j += 256) {
        int h   = j & 15;
        int pix = (j >> 4) & 63;
        int grp = (j >> 10) & 1;
        int ks  = j >> 11;
        int k = ks * 32 + grp * 8 + ((h < 8) ? h : (h + 8));
        _Float16 bv = (_Float16)0.0f;
        if ((K_TOT % 32 == 0) || (k < K_TOT)) {
            int ci = k / 9, r = k - ci * 9;
            int kh = r / 3, kw = r - kh * 3;
            int prow = pix >> 3, pcol = pix & 7;
            bv = (_Float16)lds_in[ci * 100 + (prow + kh) * 10 + (pcol + kw)];
        }
        lds_bpack[j] = bv;
    }
    __syncthreads();   // after this barrier lds_in is dead; lds_out may reuse it

    // ---- Stage 3: WMMA main loop ----
    const int lane = tid & 31, wave = tid >> 5;
    const int wm = wave & 3, wn = wave >> 2;   // wm: cout block, wn: pixel half
    const int n = lane & 15, grp = lane >> 4;

    v8f acc0 = {};
    v8f acc1 = {};
    const _Float16* aBase = packA + (size_t)(wm * KST) * 512 + (size_t)lane * 16;
    const _Float16* bBase0 = lds_bpack + ((size_t)grp * 64 + wn * 32 + n) * 16;

#pragma unroll 2
    for (int ks = 0; ks < KST; ++ks) {
        // A fragment: 32 contiguous bytes per lane from packed weights (L2 hot).
        v16h afrag = *(const v16h*)(aBase + (size_t)ks * 512);
        // B fragments: single 32B vector LDS read each.
        v16h bfrag0 = *(const v16h*)(bBase0 + (size_t)ks * 2048);
        v16h bfrag1 = *(const v16h*)(bBase0 + (size_t)ks * 2048 + 16 * 16);
        acc0 = __builtin_amdgcn_wmma_f32_16x16x32_f16(
            false, afrag, false, bfrag0, (short)0, acc0, false, false);
        acc1 = __builtin_amdgcn_wmma_f32_16x16x32_f16(
            false, afrag, false, bfrag1, (short)0, acc1, false, false);
    }

    // ---- Epilogue: bias (+BN +ReLU), scatter to LDS [cout][pixel] ----
    const float inv = 0.99999500003749969f;  // 1/sqrt(1+1e-5)
#pragma unroll
    for (int v = 0; v < 8; ++v) {
        int cout = wm * 16 + v + grp * 8;    // C/D layout: M = v + 8*(lane/16)
        float sc = BNRELU ? (g[cout] * inv) : 1.0f;
        float bt = BNRELU ? be[cout] : 0.0f;
        float a0 = acc0[v] + cb[cout];
        float a1 = acc1[v] + cb[cout];
        if (BNRELU) {
            a0 = fmaxf(a0 * sc + bt, 0.0f);
            a1 = fmaxf(a1 * sc + bt, 0.0f);
        }
        lds_out[cout * 64 + wn * 32 + n]      = a0;
        lds_out[cout * 64 + wn * 32 + 16 + n] = a1;
    }
    __syncthreads();

    // ---- Stage 4: store (pooled or direct) ----
    if (POOL) {
        const int Hp = Hc / 2, Wp = Wc / 2;
        // 64 couts x (4x4) pooled outputs = 1024 values, 4 per thread.
        for (int i = tid; i < COUT * 16; i += 256) {
            int c  = i >> 4;
            int pr = (i >> 2) & 3;
            int pc = i & 3;
            int p0 = (pr * 2) * 8 + pc * 2;
            const float* row = &lds_out[c * 64];
            float m = fmaxf(fmaxf(row[p0], row[p0 + 1]),
                            fmaxf(row[p0 + 8], row[p0 + 9]));
            int ph = (oh0 >> 1) + pr, pw = (ow0 >> 1) + pc;
            if (ph < Hp && pw < Wp)
                y[((size_t)(b * COUT + c) * Hp + ph) * Wp + pw] = m;
        }
    } else {
        for (int i = tid; i < COUT * 64; i += 256) {
            int c = i >> 6, p = i & 63;
            int oh = oh0 + (p >> 3), ow = ow0 + (p & 7);
            if (oh < Hc && ow < Wc)
                y[((size_t)(b * COUT + c) * Hc + oh) * Wc + ow] = lds_out[c * 64 + p];
        }
    }
}

// ---------------------------------------------------------------------------
// Attention score: per-pixel MLP(64->32->1), softmax over H*W, then
// attsum = 3x3-ones conv of normalized attention. One block per image.
// ---------------------------------------------------------------------------
__global__ void att_score_kernel(const float* __restrict__ ac,
                                 const float* __restrict__ w1,
                                 const float* __restrict__ b1,
                                 const float* __restrict__ w2,
                                 const float* __restrict__ b2,
                                 float* __restrict__ attsum,
                                 int H, int W) {
    const int b = blockIdx.x, tid = threadIdx.x;
    const int HW = H * W;
    __shared__ float s_s[736];   // H*W <= 729
    __shared__ float s_red[256];
    const float* acb = ac + (size_t)b * COUT * HW;

    for (int p = tid; p < HW; p += 256) {
        float hbuf[32];
#pragma unroll
        for (int j = 0; j < 32; ++j) hbuf[j] = b1[j];
        for (int c = 0; c < COUT; ++c) {
            float a = acb[c * HW + p];
#pragma unroll
            for (int j = 0; j < 32; ++j) hbuf[j] += a * w1[c * 32 + j];
        }
        float sc = b2[0];
#pragma unroll
        for (int j = 0; j < 32; ++j) sc += fmaxf(hbuf[j], 0.0f) * w2[j];
        s_s[p] = sc;
    }
    __syncthreads();

    // max reduction
    float lm = -INFINITY;
    for (int p = tid; p < HW; p += 256) lm = fmaxf(lm, s_s[p]);
    s_red[tid] = lm;
    __syncthreads();
    for (int st = 128; st > 0; st >>= 1) {
        if (tid < st) s_red[tid] = fmaxf(s_red[tid], s_red[tid + st]);
        __syncthreads();
    }
    float mx = s_red[0];
    __syncthreads();

    // exp + sum reduction
    float ls = 0.0f;
    for (int p = tid; p < HW; p += 256) {
        float e = expf(s_s[p] - mx);
        s_s[p] = e;
        ls += e;
    }
    s_red[tid] = ls;
    __syncthreads();
    for (int st = 128; st > 0; st >>= 1) {
        if (tid < st) s_red[tid] += s_red[tid + st];
        __syncthreads();
    }
    const float invT = 1.0f / s_red[0];
    __syncthreads();

    // attsum = 3x3-ones stencil of normalized attention (zero padded)
    for (int p = tid; p < HW; p += 256) {
        int h = p / W, w = p - h * W;
        float a = 0.0f;
#pragma unroll
        for (int dh = -1; dh <= 1; ++dh) {
#pragma unroll
            for (int dw = -1; dw <= 1; ++dw) {
                int hh = h + dh, ww = w + dw;
                if (hh >= 0 && hh < H && ww >= 0 && ww < W) a += s_s[hh * W + ww];
            }
        }
        attsum[(size_t)b * HW + p] = a * invT;
    }
}

// ---------------------------------------------------------------------------
// out[b, colOff + c] = sum_hw feat[b,c,h,w] * attsum[b,h,w]
// One block per (b, c).
// ---------------------------------------------------------------------------
__global__ void att_out_kernel(const float* __restrict__ feat,
                               const float* __restrict__ attsum,
                               float* __restrict__ out,
                               int HW, int colOff) {
    const int b = blockIdx.x >> 6;
    const int c = blockIdx.x & 63;
    const int tid = threadIdx.x;
    __shared__ float red[256];
    const float* f = feat + ((size_t)b * COUT + c) * HW;
    const float* a = attsum + (size_t)b * HW;
    float s = 0.0f;
    for (int p = tid; p < HW; p += 256) s += f[p] * a[p];
    red[tid] = s;
    __syncthreads();
    for (int st = 128; st > 0; st >>= 1) {
        if (tid < st) red[tid] += red[tid + st];
        __syncthreads();
    }
    if (tid == 0) out[b * 128 + colOff + c] = red[0];
}

// ---------------------------------------------------------------------------
static inline int conv_grid(int H, int W, int pad) {
    int Hc = H + 2 * pad - 2, Wc = W + 2 * pad - 2;
    int nbh = (Hc + 7) / 8, nbw = (Wc + 7) / 8;
    return BATCH * nbh * nbw;
}

extern "C" void kernel_launch(void* const* d_in, const int* in_sizes, int n_in,
                              void* d_out, int out_size, void* d_ws, size_t ws_size,
                              hipStream_t stream) {
    (void)in_sizes; (void)n_in; (void)out_size; (void)ws_size;
    const float* image = (const float*)d_in[0];
    const float* c1w = (const float*)d_in[1];  const float* c1b = (const float*)d_in[2];
    const float* g1  = (const float*)d_in[3];  const float* be1 = (const float*)d_in[4];
    const float* c2w = (const float*)d_in[5];  const float* c2b = (const float*)d_in[6];
    const float* g2  = (const float*)d_in[7];  const float* be2 = (const float*)d_in[8];
    const float* c3w = (const float*)d_in[9];  const float* c3b = (const float*)d_in[10];
    const float* g3  = (const float*)d_in[11]; const float* be3 = (const float*)d_in[12];
    const float* c4w = (const float*)d_in[13]; const float* c4b = (const float*)d_in[14];
    const float* g4  = (const float*)d_in[15]; const float* be4 = (const float*)d_in[16];
    const float* a1cw = (const float*)d_in[17]; const float* a1cb = (const float*)d_in[18];
    const float* a1w1 = (const float*)d_in[19]; const float* a1b1 = (const float*)d_in[20];
    const float* a1w2 = (const float*)d_in[21]; const float* a1b2 = (const float*)d_in[22];
    const float* a2cw = (const float*)d_in[23]; const float* a2cb = (const float*)d_in[24];
    const float* a2w1 = (const float*)d_in[25]; const float* a2b1 = (const float*)d_in[26];
    const float* a2w2 = (const float*)d_in[27]; const float* a2b2 = (const float*)d_in[28];
    float* out = (float*)d_out;

    // Workspace carving (256B aligned)
    char* p = (char*)d_ws;
    auto carve = [&](size_t bytes) -> void* {
        void* r = (void*)p;
        p += (bytes + 255) & ~(size_t)255;
        return r;
    };
    const size_t o1_e = (size_t)BATCH * COUT * 111 * 111;
    const size_t o2_e = (size_t)BATCH * COUT * 54 * 54;
    const size_t o3_e = (size_t)BATCH * COUT * 27 * 27;
    const size_t o4_e = (size_t)BATCH * COUT * 13 * 13;
    float* o1  = (float*)carve(o1_e * 4);
    float* o2  = (float*)carve(o2_e * 4);
    float* o3  = (float*)carve(o3_e * 4);
    float* o4  = (float*)carve(o4_e * 4);
    float* ac1 = (float*)carve(o3_e * 4);
    float* ac2 = (float*)carve(o4_e * 4);
    float* as1 = (float*)carve((size_t)BATCH * 729 * 4);
    float* as2 = (float*)carve((size_t)BATCH * 169 * 4);
    const int KST1 = 1, KST64 = 18;            // ceil(27/32), ceil(576/32)
    _Float16* pk1  = (_Float16*)carve((size_t)4 * KST1 * 512 * 2);
    _Float16* pk2  = (_Float16*)carve((size_t)4 * KST64 * 512 * 2);
    _Float16* pk3  = (_Float16*)carve((size_t)4 * KST64 * 512 * 2);
    _Float16* pk4  = (_Float16*)carve((size_t)4 * KST64 * 512 * 2);
    _Float16* pka1 = (_Float16*)carve((size_t)4 * KST64 * 512 * 2);
    _Float16* pka2 = (_Float16*)carve((size_t)4 * KST64 * 512 * 2);

    // 1) Pack weights to WMMA fragment order
    {
        int t1 = 4 * KST1 * 512, t64 = 4 * KST64 * 512;
        pack_w_kernel<3><<<(t1 + 255) / 256, 256, 0, stream>>>(c1w, pk1);
        pack_w_kernel<64><<<(t64 + 255) / 256, 256, 0, stream>>>(c2w, pk2);
        pack_w_kernel<64><<<(t64 + 255) / 256, 256, 0, stream>>>(c3w, pk3);
        pack_w_kernel<64><<<(t64 + 255) / 256, 256, 0, stream>>>(c4w, pk4);
        pack_w_kernel<64><<<(t64 + 255) / 256, 256, 0, stream>>>(a1cw, pka1);
        pack_w_kernel<64><<<(t64 + 255) / 256, 256, 0, stream>>>(a2cw, pka2);
    }

    // 2) Fused conv blocks
    conv_block_kernel<3, 0, true, true><<<conv_grid(224, 224, 0), 256, 0, stream>>>(
        image, pk1, c1b, g1, be1, o1, 224, 224);
    conv_block_kernel<64, 0, true, true><<<conv_grid(111, 111, 0), 256, 0, stream>>>(
        o1, pk2, c2b, g2, be2, o2, 111, 111);
    conv_block_kernel<64, 1, true, true><<<conv_grid(54, 54, 1), 256, 0, stream>>>(
        o2, pk3, c3b, g3, be3, o3, 54, 54);

    // 3) Attention branch 1 on o3 (27x27)
    conv_block_kernel<64, 1, false, false><<<conv_grid(27, 27, 1), 256, 0, stream>>>(
        o3, pka1, a1cb, nullptr, nullptr, ac1, 27, 27);
    att_score_kernel<<<BATCH, 256, 0, stream>>>(ac1, a1w1, a1b1, a1w2, a1b2, as1, 27, 27);
    att_out_kernel<<<BATCH * COUT, 256, 0, stream>>>(o3, as1, out, 27 * 27, 0);

    // 4) Block 4 + attention branch 2 on o4 (13x13)
    conv_block_kernel<64, 1, true, true><<<conv_grid(27, 27, 1), 256, 0, stream>>>(
        o3, pk4, c4b, g4, be4, o4, 27, 27);
    conv_block_kernel<64, 1, false, false><<<conv_grid(13, 13, 1), 256, 0, stream>>>(
        o4, pka2, a2cb, nullptr, nullptr, ac2, 13, 13);
    att_score_kernel<<<BATCH, 256, 0, stream>>>(ac2, a2w1, a2b1, a2w2, a2b2, as2, 13, 13);
    att_out_kernel<<<BATCH * COUT, 256, 0, stream>>>(o4, as2, out, 13 * 13, 64);
}
